// AxialMultiHeadAttention_3126736192166
// MI455X (gfx1250) — compile-verified, matrix operand order checked by reading, not run on hardware
//
#include <hip/hip_runtime.h>
#include <cstddef>

// ---------------------------------------------------------------------------
// AxialMultiHeadAttention (temporal MHA -> social MHA) for gfx1250 (MI455X)
// All GEMMs use v_wmma_f32_16x16x32_bf16; softmax fused in registers.
// ---------------------------------------------------------------------------

#define Bb  8
#define Ll  128
#define Nt  64
#define Dm  256
#define Hh  8
#define Dk  32
#define Ttok (Bb*Ll*Nt)        // 65536 tokens

typedef __attribute__((ext_vector_type(16))) __bf16 v16bf;
typedef __attribute__((ext_vector_type(8)))  float  v8f;

union FragB {
    v16bf v;
    unsigned u[8];
    __bf16 h[16];
};

__device__ __forceinline__ v8f wmma_bf16(v16bf a, v16bf b, v8f c) {
    return __builtin_amdgcn_wmma_f32_16x16x32_bf16(false, a, false, b,
                                                   (short)0, c, false, false);
}

// K index of the first element of 32-bit pair i (0..7) for this half-wave.
// 16-bit A/B fragment layout (CDNA5 ISA 7.12.2): lanes 0-15 hold K 0-7 (v0-3)
// and K 16-23 (v4-7); lanes 16-31 hold K 8-15 and K 24-31.
__device__ __forceinline__ int kmap(int i, int hi) {
    return (i < 4 ? 2 * i : 16 + 2 * (i - 4)) + (hi ? 8 : 0);
}

__device__ __forceinline__ unsigned short f2bf(float f) {
    union { __bf16 h; unsigned short s; } u;
    u.h = (__bf16)f;
    return u.s;
}

// ---------------------------------------------------------------------------
// Weight conversion: 8 x [256x256] f32 -> bf16, packed consecutively.
// ---------------------------------------------------------------------------
__global__ void convw_kernel(const float* a0, const float* a1, const float* a2,
                             const float* a3, const float* a4, const float* a5,
                             const float* a6, const float* a7,
                             unsigned short* out) {
    unsigned i = blockIdx.x * 256u + threadIdx.x;   // 0 .. 524287
    unsigned m = i >> 16, r = i & 0xFFFFu;
    const float* p;
    switch (m) {
        case 0: p = a0; break; case 1: p = a1; break;
        case 2: p = a2; break; case 3: p = a3; break;
        case 4: p = a4; break; case 5: p = a5; break;
        case 6: p = a6; break; default: p = a7; break;
    }
    out[i] = f2bf(p[r]);
}

// ---------------------------------------------------------------------------
// Input projection: Y = X @ W^T, written per-head.
//   MODE 0 (temporal): group g = b*Nt+n, rows s = l (stride Nt*Dm in X)
//   MODE 1 (social):   group g = b*Ll+l, rows s = n (stride Dm in X)
//   INF32: input is f32 [B,L,N,D]; else bf16 [T,256]
//   TRANS: output [head, Dk, Sg] (for V) else [head, Sg, Dk]
// One wave computes a 16x16 output tile; 8 WMMAs (K=256).
// ---------------------------------------------------------------------------
template<int MODE, int INF32, int TRANS>
__global__ __launch_bounds__(256) void proj_kernel(const void* __restrict__ inp,
                                                   const unsigned short* __restrict__ Wb,
                                                   unsigned short* __restrict__ outp) {
    constexpr int Sg = (MODE == 0) ? Ll : Nt;
    const int wid  = blockIdx.x * 8 + (threadIdx.x >> 5);
    const int lane = threadIdx.x & 31;
    const int ln = lane & 15, hi = lane >> 4;

    const int colTile = wid & 15;
    const int rowTile = wid >> 4;
    const int g  = rowTile / (Sg / 16);
    const int s0 = (rowTile % (Sg / 16)) * 16;
    const int s  = s0 + ln;               // A-matrix row for this lane

    size_t rowbase;
    if (MODE == 0) {
        int b = g / Nt, n = g % Nt;
        rowbase = (((size_t)b * Ll + s) * Nt + n) * Dm;
    } else {
        rowbase = ((size_t)g * Nt + s) * Dm;
    }

    const int o = colTile * 16 + ln;      // output feature (B column)
    v8f acc = {0.f,0.f,0.f,0.f,0.f,0.f,0.f,0.f};

    #pragma unroll
    for (int kk = 0; kk < 8; ++kk) {
        FragB a, bw;
        if (INF32) {
            const float* pf = (const float*)inp + rowbase + kk * 32;
            #pragma unroll
            for (int i = 0; i < 8; ++i) {
                int K = kmap(i, hi);
                a.h[2*i]   = (__bf16)pf[K];
                a.h[2*i+1] = (__bf16)pf[K + 1];
            }
        } else {
            const unsigned* pb =
                (const unsigned*)((const unsigned short*)inp + rowbase + kk * 32);
            #pragma unroll
            for (int i = 0; i < 8; ++i) a.u[i] = pb[kmap(i, hi) >> 1];
        }
        const unsigned* pw = (const unsigned*)(Wb + (size_t)o * Dm + kk * 32);
        #pragma unroll
        for (int i = 0; i < 8; ++i) bw.u[i] = pw[kmap(i, hi) >> 1];

        acc = wmma_bf16(a.v, bw.v, acc);
    }

    const int h = o >> 5, dp = o & 31;
    #pragma unroll
    for (int j = 0; j < 8; ++j) {
        int m  = j + hi * 8;
        int ss = s0 + m;
        size_t idx;
        if (!TRANS) idx = (((size_t)g * Hh + h) * Sg + ss) * Dk + dp;
        else        idx = (((size_t)g * Hh + h) * Dk + dp) * Sg + ss;
        outp[idx] = f2bf(acc[j]);
    }
}

// ---------------------------------------------------------------------------
// Fused attention: per head, scores + softmax + P@V.
//   S = sequence length (128 temporal, 64 social); one wave per 16 query rows.
//   Q,K: [head, S, Dk] bf16; Vt: [head, Dk, S] bf16 (transposed).
//   Output scattered bf16 into [B,L,N,D] token order at the head's columns.
// ---------------------------------------------------------------------------
template<int S, int MODE>
__global__ __launch_bounds__(256) void attn_kernel(const unsigned short* __restrict__ Qb,
                                                   const unsigned short* __restrict__ Kb,
                                                   const unsigned short* __restrict__ Vtb,
                                                   unsigned short* __restrict__ Cb) {
    constexpr int TILES = S / 16;
    __shared__ unsigned short ldsP[TILES * 16 * S];

    const int hid  = blockIdx.x;
    const int tid  = threadIdx.x;
    const int w    = tid >> 5;
    const int lane = tid & 31;
    const int ln = lane & 15, hi = lane >> 4;

    const size_t headOff = (size_t)hid * S * Dk;
    const unsigned short* q  = Qb  + headOff;
    const unsigned short* k  = Kb  + headOff;
    const unsigned short* vt = Vtb + headOff;

    size_t outBase, outRS;
    if (MODE == 0) {            // temporal: hid = (b*Nt+n)*Hh + h, rows = l
        int bn = hid / Hh, h = hid % Hh;
        int b = bn / Nt, n = bn % Nt;
        outBase = (((size_t)b * Ll) * Nt + n) * Dm + (size_t)h * Dk;
        outRS   = (size_t)Nt * Dm;
    } else {                    // social: hid = (b*Ll+l)*Hh + h, rows = n
        int bl = hid / Hh, h = hid % Hh;
        outBase = ((size_t)bl * Nt) * Dm + (size_t)h * Dk;
        outRS   = Dm;
    }

    // --- A fragment: this wave's 16 query rows (K = Dk = 32 fits one frag) ---
    FragB aq;
    {
        const unsigned* p = (const unsigned*)(q + (size_t)(w * 16 + ln) * Dk);
        #pragma unroll
        for (int i = 0; i < 8; ++i) aq.u[i] = p[kmap(i, hi) >> 1];
    }

    // --- scores S = Q K^T ---
    v8f acc[TILES];
    #pragma unroll
    for (int ct = 0; ct < TILES; ++ct) {
        FragB bk;
        const unsigned* p = (const unsigned*)(k + (size_t)(ct * 16 + ln) * Dk);
        #pragma unroll
        for (int i = 0; i < 8; ++i) bk.u[i] = p[kmap(i, hi) >> 1];
        v8f z = {0.f,0.f,0.f,0.f,0.f,0.f,0.f,0.f};
        acc[ct] = wmma_bf16(aq.v, bk.v, z);
    }

    // --- softmax over the S-wide rows (rows live in 16-lane halves) ---
    const float scale = 0.17677669529663688f;   // 1/sqrt(32)
    #pragma unroll
    for (int j = 0; j < 8; ++j) {
        float mx = -3.4e38f;
        #pragma unroll
        for (int ct = 0; ct < TILES; ++ct) mx = fmaxf(mx, acc[ct][j]);
        #pragma unroll
        for (int d = 1; d < 16; d <<= 1) mx = fmaxf(mx, __shfl_xor(mx, d, 32));
        float sum = 0.f;
        #pragma unroll
        for (int ct = 0; ct < TILES; ++ct) {
            float e = __expf((acc[ct][j] - mx) * scale);
            acc[ct][j] = e;
            sum += e;
        }
        #pragma unroll
        for (int d = 1; d < 16; d <<= 1) sum += __shfl_xor(sum, d, 32);
        float inv = 1.0f / sum;
        #pragma unroll
        for (int ct = 0; ct < TILES; ++ct) acc[ct][j] *= inv;
    }

    // --- stage P (bf16) through LDS: C-layout -> A-layout transpose ---
    unsigned short* myP = ldsP + w * 16 * S;
    #pragma unroll
    for (int ct = 0; ct < TILES; ++ct) {
        #pragma unroll
        for (int j = 0; j < 8; ++j) {
            int m = j + hi * 8;
            int c = ct * 16 + ln;
            myP[m * S + c] = f2bf(acc[ct][j]);
        }
    }
    __syncthreads();

    // --- O = P @ V  (N tiles: 2 of 16 dk-cols; K tiles: S/32) ---
    #pragma unroll
    for (int nt = 0; nt < 2; ++nt) {
        v8f o = {0.f,0.f,0.f,0.f,0.f,0.f,0.f,0.f};
        #pragma unroll
        for (int kt = 0; kt < S / 32; ++kt) {
            FragB ap, bv;
            const unsigned* pp = (const unsigned*)(myP + (size_t)ln * S) + kt * 16;
            #pragma unroll
            for (int i = 0; i < 8; ++i) ap.u[i] = pp[kmap(i, hi) >> 1];
            const unsigned* pv =
                (const unsigned*)(vt + (size_t)(nt * 16 + ln) * S) + kt * 16;
            #pragma unroll
            for (int i = 0; i < 8; ++i) bv.u[i] = pv[kmap(i, hi) >> 1];
            o = wmma_bf16(ap.v, bv.v, o);
        }
        #pragma unroll
        for (int j = 0; j < 8; ++j) {
            int m = j + hi * 8;
            int r = w * 16 + m;
            int c = nt * 16 + ln;
            Cb[outBase + (size_t)r * outRS + c] = f2bf(o[j]);
        }
    }
}

// ---------------------------------------------------------------------------
// Output projection: Y = X @ Wd^T + bias.  X is bf16 [T,256] token-major.
//   OUTF32: write f32 (final d_out) else bf16 (t_out feeding social stage).
// ---------------------------------------------------------------------------
template<int OUTF32>
__global__ __launch_bounds__(256) void oproj_kernel(const unsigned short* __restrict__ X,
                                                    const unsigned short* __restrict__ Wb,
                                                    const float* __restrict__ bias,
                                                    void* __restrict__ outp) {
    const int wid  = blockIdx.x * 8 + (threadIdx.x >> 5);
    const int lane = threadIdx.x & 31;
    const int ln = lane & 15, hi = lane >> 4;

    const int colTile = wid & 15;
    const int rowTile = wid >> 4;
    const int t0 = rowTile * 16;
    const int o  = colTile * 16 + ln;

    v8f acc = {0.f,0.f,0.f,0.f,0.f,0.f,0.f,0.f};
    const unsigned* px = (const unsigned*)(X  + (size_t)(t0 + ln) * Dm);
    const unsigned* pw = (const unsigned*)(Wb + (size_t)o * Dm);

    #pragma unroll
    for (int kk = 0; kk < 8; ++kk) {
        FragB a, bw;
        #pragma unroll
        for (int i = 0; i < 8; ++i) {
            int wi = kk * 16 + (kmap(i, hi) >> 1);
            a.u[i]  = px[wi];
            bw.u[i] = pw[wi];
        }
        acc = wmma_bf16(a.v, bw.v, acc);
    }

    float bo = bias[o];
    #pragma unroll
    for (int j = 0; j < 8; ++j) {
        int m = j + hi * 8;
        size_t idx = (size_t)(t0 + m) * Dm + o;
        if (OUTF32) ((float*)outp)[idx] = acc[j] + bo;
        else ((unsigned short*)outp)[idx] = f2bf(acc[j] + bo);
    }
}

// ---------------------------------------------------------------------------
extern "C" void kernel_launch(void* const* d_in, const int* in_sizes, int n_in,
                              void* d_out, int out_size, void* d_ws, size_t ws_size,
                              hipStream_t stream) {
    const float* q    = (const float*)d_in[0];
    const float* key  = (const float*)d_in[1];
    const float* val  = (const float*)d_in[2];
    const float* t_wq = (const float*)d_in[3];
    const float* t_wk = (const float*)d_in[4];
    const float* t_wv = (const float*)d_in[5];
    const float* t_wd = (const float*)d_in[6];
    const float* t_bd = (const float*)d_in[7];
    const float* s_wq = (const float*)d_in[8];
    const float* s_wk = (const float*)d_in[9];
    const float* s_wv = (const float*)d_in[10];
    const float* s_wd = (const float*)d_in[11];
    const float* s_bd = (const float*)d_in[12];

    const size_t WSZ = (size_t)Dm * Dm;           // 65536 per weight
    const size_t HB  = (size_t)Ttok * Dm;         // 16,777,216 elems per buffer
    const size_t need = (8 * WSZ + 5 * HB) * sizeof(unsigned short);
    if (ws_size < need) return;                    // deterministic no-op guard

    unsigned short* Wbf  = (unsigned short*)d_ws;
    unsigned short* Qb   = Wbf + 8 * WSZ;
    unsigned short* Kb   = Qb + HB;
    unsigned short* Vb   = Kb + HB;
    unsigned short* Cbuf = Vb + HB;
    unsigned short* Tout = Cbuf + HB;

    const unsigned short* Wtq = Wbf + 0 * WSZ;
    const unsigned short* Wtk = Wbf + 1 * WSZ;
    const unsigned short* Wtv = Wbf + 2 * WSZ;
    const unsigned short* Wtd = Wbf + 3 * WSZ;
    const unsigned short* Wsq = Wbf + 4 * WSZ;
    const unsigned short* Wsk = Wbf + 5 * WSZ;
    const unsigned short* Wsv = Wbf + 6 * WSZ;
    const unsigned short* Wsd = Wbf + 7 * WSZ;

    // 0) weights f32 -> bf16
    convw_kernel<<<2048, 256, 0, stream>>>(t_wq, t_wk, t_wv, t_wd,
                                           s_wq, s_wk, s_wv, s_wd, Wbf);

    // 1) temporal projections (rows = l per (b,n)); V transposed
    proj_kernel<0, 1, 0><<<8192, 256, 0, stream>>>(q,   Wtq, Qb);
    proj_kernel<0, 1, 0><<<8192, 256, 0, stream>>>(key, Wtk, Kb);
    proj_kernel<0, 1, 1><<<8192, 256, 0, stream>>>(val, Wtv, Vb);

    // 2) temporal attention over L (4096 heads, 8 waves each)
    attn_kernel<128, 0><<<Bb * Nt * Hh, 256, 0, stream>>>(Qb, Kb, Vb, Cbuf);

    // 3) temporal output projection -> t_out (bf16, token-major)
    oproj_kernel<0><<<8192, 256, 0, stream>>>(Cbuf, Wtd, t_bd, (void*)Tout);

    // 4) social projections (rows = n per (b,l)); Q from t_out (bf16)
    proj_kernel<1, 0, 0><<<8192, 256, 0, stream>>>(Tout, Wsq, Qb);
    proj_kernel<1, 1, 0><<<8192, 256, 0, stream>>>(key,  Wsk, Kb);
    proj_kernel<1, 1, 1><<<8192, 256, 0, stream>>>(val,  Wsv, Vb);

    // 5) social attention over N (8192 heads, 4 waves each)
    attn_kernel<64, 1><<<Bb * Ll * Hh, 128, 0, stream>>>(Qb, Kb, Vb, Cbuf);

    // 6) final output projection -> f32 d_out
    oproj_kernel<1><<<8192, 256, 0, stream>>>(Cbuf, Wsd, s_bd, d_out);
}
